// SWSAttention_78572131713413
// MI455X (gfx1250) — compile-verified, matrix operand order checked by reading, not run on hardware
//
#include <hip/hip_runtime.h>

// ---------- types ----------
typedef __bf16 bf16_t;
typedef __attribute__((ext_vector_type(16))) __bf16 v16bf;
typedef __attribute__((ext_vector_type(8)))  __bf16 v8bf;
typedef __attribute__((ext_vector_type(8)))  float  v8f;

// Problem constants
#define BB 4
#define CC 96
#define HH 256
#define WW2 256
#define NHEAD 4
#define DHEAD 24
#define DPAD 32
#define NWIN 64      // windows per image
#define SNW 64       // sub-positions per window
#define NPIX 16      // pixels per sub-block
#define KEYS 1024    // NWIN*NPIX
#define CHUNK 128
#define NCHUNK 8
#define SCALE_Q 0.20412414523193153f   // 24^-0.5
#define NEG_BIG -3.0e38f

// ---------- CDNA5 async-to-LDS (ASYNCcnt) helpers ----------
#if __has_builtin(__builtin_amdgcn_s_wait_asynccnt)
#define WAIT_ASYNC() __builtin_amdgcn_s_wait_asynccnt(0)
#else
#define WAIT_ASYNC() asm volatile("s_wait_asynccnt 0x0" ::: "memory")
#endif

// Per-lane scattered DMA: global (fp32 element) -> LDS byte address.
// vdst = 32-bit LDS byte offset VGPR, vaddr = 64-bit global address VGPR pair.
__device__ __forceinline__ void async_load_b32(unsigned int lds_byte_addr,
                                               const float* gaddr) {
    asm volatile("global_load_async_to_lds_b32 %0, %1, off"
                 :: "v"(lds_byte_addr), "v"(gaddr)
                 : "memory");
}

// ---------- WMMA fragment helpers (16x16x32 bf16, wave32) ----------
// A (16x32, row-major, leading dim ld): lane L(<16) row M=L holds K {k0..k0+7, k0+16..k0+23}
// as two contiguous 8-elem (16B) segments; lanes 16..31 hold the {+8, +24} halves.
// B is loaded identically from row-major B^T (i.e. the weight matrix itself).
__device__ __forceinline__ v16bf load_frag(const bf16_t* base, int ld, int lane) {
    const int r = lane & 15, half = lane >> 4;
    const bf16_t* p = base + r * ld + 8 * half;
    union { v16bf v; v8bf h[2]; } u;
    u.h[0] = *(const v8bf*)(p);
    u.h[1] = *(const v8bf*)(p + 16);
    return u.v;
}

__device__ __forceinline__ v8f wmma_bf16(v16bf a, v16bf b, v8f c) {
    return __builtin_amdgcn_wmma_f32_16x16x32_bf16(false, a, false, b, (short)0, c, false, false);
}

// ---------- kernel 0: weight conversion + zero padded-q buffer ----------
__global__ void k0_prep(const float* __restrict__ qw, const float* __restrict__ kvw,
                        const float* __restrict__ pw,
                        bf16_t* __restrict__ qwb, bf16_t* __restrict__ kvwb,
                        bf16_t* __restrict__ pwb, unsigned int* __restrict__ qws_u32) {
    const int i = blockIdx.x * blockDim.x + threadIdx.x;
    if (i < 9216)            qwb[i]         = (bf16_t)qw[i];
    else if (i < 9216+18432) kvwb[i - 9216] = (bf16_t)kvw[i - 9216];
    else if (i < 36864)      pwb[i - 27648] = (bf16_t)pw[i - 27648];
    if (i < 1048576)         qws_u32[i] = 0u;   // zero 4 MB padded-q buffer
}

// ---------- kernel 1: pooled[b,s,w,c] = mean over 16 pixels (coalesced) ----------
// block = ((b*96 + c)*8 + wh)*8 + shi ; 64 threads: t = ww*8 + swi
// Lanes read fully contiguous rows of x (float4 per lane, 1 KB per row per block).
__global__ void k1_pool(const float* __restrict__ x, bf16_t* __restrict__ pooled) {
    const int blk = blockIdx.x;
    const int shi = blk & 7, wh = (blk >> 3) & 7;
    const int c   = (blk >> 6) % CC, b = blk / (CC * 64);
    const int t   = threadIdx.x;              // 0..63
    const int ww  = t >> 3, swi = t & 7;
    const size_t rowbase = ((size_t)(b * CC + c) * HH + wh * 32 + shi * 4) * WW2;
    const int wc = ww * 32 + swi * 4;
    float sum = 0.f;
#pragma unroll
    for (int ph = 0; ph < 4; ++ph) {
        const float4 v = *(const float4*)(x + rowbase + (size_t)ph * WW2 + wc);
        sum += v.x + v.y + v.z + v.w;
    }
    const int s = shi * 8 + swi, w = wh * 8 + ww;
    pooled[(size_t)((b * 64 + s) * 64 + w) * CC + c] = (bf16_t)(sum * (1.0f / 16.0f));
}

// ---------- kernel 2: q = (pooled @ q_w^T + q_b) * scale, head-padded bf16 ----------
__global__ __launch_bounds__(256) void k2_qproj(const bf16_t* __restrict__ pooled,
                                                const bf16_t* __restrict__ qwb,
                                                const float* __restrict__ q_b,
                                                bf16_t* __restrict__ q_ws) {
    const int wave = threadIdx.x >> 5, lane = threadIdx.x & 31;
    const int lr = lane & 15, lh = lane >> 4;
    const int mt = blockIdx.x * 8 + wave;            // 0..1023 M-tiles (16384 rows)
    const bf16_t* abase = pooled + (size_t)mt * 16 * CC;
#pragma unroll
    for (int nt = 0; nt < 6; ++nt) {
        v8f acc;
#pragma unroll
        for (int j = 0; j < 8; ++j) acc[j] = 0.f;
#pragma unroll
        for (int ks = 0; ks < 3; ++ks) {
            v16bf a = load_frag(abase + ks * 32, CC, lane);
            v16bf b = load_frag(qwb + nt * 16 * CC + ks * 32, CC, lane);
            acc = wmma_bf16(a, b, acc);
        }
        const int c = nt * 16 + lr;
        const float bias = q_b[c];
        const int off = (c / DHEAD) * DPAD + (c % DHEAD);
#pragma unroll
        for (int r = 0; r < 8; ++r) {
            const int row = mt * 16 + r + 8 * lh;
            q_ws[(size_t)row * 128 + off] = (bf16_t)((acc[r] + bias) * SCALE_Q);
        }
    }
}

// ---------- kernel 3: fused kv-proj + flash attention + out proj + scatter ----------
__global__ __launch_bounds__(256) void k3_attn(const float* __restrict__ x,
                                               const float* __restrict__ kv_b,
                                               const float* __restrict__ proj_b,
                                               const bf16_t* __restrict__ q_ws,
                                               const bf16_t* __restrict__ kvwb,
                                               const bf16_t* __restrict__ pwb,
                                               float* __restrict__ out) {
    const int bs = blockIdx.x;               // 0..255 = (b, s)
    const int b = bs >> 6, s = bs & 63;
    const int shi = s >> 3, swi = s & 7;
    const int tid = threadIdx.x, wave = tid >> 5, lane = tid & 31;
    const int lr = lane & 15, lh = lane >> 4;

    __shared__ float  stage[CHUNK * CC];      // async landing buffer (fp32)   48 KB
    __shared__ bf16_t q_lds[64 * 128];        // queries, head-padded          16 KB
    __shared__ bf16_t mchunk[CHUNK * CC];     // gathered pixels (A of kv)     24 KB
    __shared__ bf16_t kpad[CHUNK * 128];      // keys, [key][head*32+d] padded 32 KB
    __shared__ bf16_t vT[NHEAD * DPAD * CHUNK]; // values, [h][d][key]         32 KB
    __shared__ float  scores[64 * CHUNK];     //                               32 KB
    __shared__ bf16_t p_lds[64 * CHUNK];      // softmax probs (A of AV)       16 KB
    __shared__ float  m_i[NHEAD][64], l_i[NHEAD][64], alpha_s[64];

    const size_t xb = (size_t)b * CC * HH * WW2;

    // global address of element i (= key*96 + c) of chunk chv
    auto gaddr_of = [&](int chv, int i) -> const float* {
        const int kk = i / CC, c = i % CC;
        const int kg = chv * CHUNK + kk;
        const int nw = kg >> 4, n = kg & 15;
        const int hh = (nw >> 3) * 32 + shi * 4 + (n >> 2);
        const int wc = (nw & 7) * 32 + swi * 4 + (n & 3);
        return x + xb + (size_t)c * (HH * WW2) + hh * WW2 + wc;
    };

    // prologue: kick off async gather of chunk 0 while we set up
    for (int i = tid; i < CHUNK * CC; i += 256)
        async_load_b32((unsigned int)(uintptr_t)&stage[i], gaddr_of(0, i));

    // init
    for (int i = tid; i < 64 * 128; i += 256)
        q_lds[i] = q_ws[(size_t)bs * 64 * 128 + i];
    for (int i = tid; i < CHUNK * NHEAD * 8; i += 256) {   // zero K-dim pad cols of kpad
        const int key = i >> 5, j = i & 31;
        kpad[key * 128 + (j >> 3) * DPAD + DHEAD + (j & 7)] = (bf16_t)0.0f;
    }
    if (tid < 64)
        for (int h = 0; h < NHEAD; ++h) { m_i[h][tid] = NEG_BIG; l_i[h][tid] = 0.f; }
    v8f o_acc[NHEAD];
#pragma unroll
    for (int h = 0; h < NHEAD; ++h)
#pragma unroll
        for (int j = 0; j < 8; ++j) o_acc[h][j] = 0.f;
    __syncthreads();

    for (int chv = 0; chv < NCHUNK; ++chv) {
        // 1) wait for this chunk's async DMA, convert fp32 -> bf16, then
        //    immediately kick off the next chunk's DMA (double buffered:
        //    it streams while the WMMAs below run).
        WAIT_ASYNC();
        __syncthreads();
        for (int i = tid; i < CHUNK * CC; i += 256)
            mchunk[i] = (bf16_t)stage[i];
        // Safe: each thread overwrites only staging elements whose ds_load it
        // already consumed (data dependence), so no cross-thread WAR hazard.
        if (chv + 1 < NCHUNK)
            for (int i = tid; i < CHUNK * CC; i += 256)
                async_load_b32((unsigned int)(uintptr_t)&stage[i], gaddr_of(chv + 1, i));
        __syncthreads();

        // 2) kv projection: (128 x 96) @ kv_w^T -> k (bf16 key-major) & v (bf16 d-major)
        for (int t = wave; t < 96; t += 8) {
            const int mt = t / 12, nt = t % 12;
            v8f acc;
#pragma unroll
            for (int j = 0; j < 8; ++j) acc[j] = 0.f;
#pragma unroll
            for (int ks = 0; ks < 3; ++ks) {
                v16bf a = load_frag(&mchunk[mt * 16 * CC + ks * 32], CC, lane);
                v16bf bb = load_frag(&kvwb[nt * 16 * CC + ks * 32], CC, lane);
                acc = wmma_bf16(a, bb, acc);
            }
            const int c = nt * 16 + lr;
            const float bias = kv_b[c];
            if (c < CC) {
                const int off = (c / DHEAD) * DPAD + (c % DHEAD);
#pragma unroll
                for (int r = 0; r < 8; ++r) {
                    const int key = mt * 16 + r + 8 * lh;
                    kpad[key * 128 + off] = (bf16_t)(acc[r] + bias);
                }
            } else {
                const int c2 = c - CC;
                const int off = ((c2 / DHEAD) * DPAD + (c2 % DHEAD)) * CHUNK;
#pragma unroll
                for (int r = 0; r < 8; ++r) {
                    const int key = mt * 16 + r + 8 * lh;
                    vT[off + key] = (bf16_t)(acc[r] + bias);
                }
            }
        }
        __syncthreads();

        // 3) per-head attention on this chunk
        for (int h = 0; h < NHEAD; ++h) {
            // QK^T : 4 m-tiles x 8 n-tiles, K=32 (24 real + zero pad)
            for (int t = wave; t < 32; t += 8) {
                const int mt = t >> 3, nt = t & 7;
                v16bf a = load_frag(&q_lds[mt * 16 * 128 + h * DPAD], 128, lane);
                v16bf bb = load_frag(&kpad[nt * 16 * 128 + h * DPAD], 128, lane);
                v8f acc;
#pragma unroll
                for (int j = 0; j < 8; ++j) acc[j] = 0.f;
                acc = wmma_bf16(a, bb, acc);
                const int key = nt * 16 + lr;
#pragma unroll
                for (int r = 0; r < 8; ++r)
                    scores[(mt * 16 + r + 8 * lh) * CHUNK + key] = acc[r];
            }
            __syncthreads();

            // online softmax (one thread per query row)
            if (tid < 64) {
                const int q = tid;
                const float mo = m_i[h][q];
                float mc = NEG_BIG;
                for (int j = 0; j < CHUNK; ++j) mc = fmaxf(mc, scores[q * CHUNK + j]);
                const float mn = fmaxf(mo, mc);
                const float al = __expf(mo - mn);
                float sum = 0.f;
                for (int j = 0; j < CHUNK; ++j) {
                    const float p = __expf(scores[q * CHUNK + j] - mn);
                    sum += p;
                    p_lds[q * CHUNK + j] = (bf16_t)p;
                }
                m_i[h][q] = mn;
                l_i[h][q] = l_i[h][q] * al + sum;
                alpha_s[q] = al;
            }
            __syncthreads();

            // AV: each wave owns one 16x16 tile of (64 x 32pad) per head, acc persists
            {
                const int mt = wave & 3, nt = wave >> 2;
#pragma unroll
                for (int r = 0; r < 8; ++r)
                    o_acc[h][r] *= alpha_s[mt * 16 + r + 8 * lh];
#pragma unroll
                for (int ks = 0; ks < 4; ++ks) {
                    v16bf a = load_frag(&p_lds[mt * 16 * CHUNK + ks * 32], CHUNK, lane);
                    v16bf bb = load_frag(&vT[(h * DPAD + nt * 16) * CHUNK + ks * 32], CHUNK, lane);
                    o_acc[h] = wmma_bf16(a, bb, o_acc[h]);
                }
            }
            __syncthreads();
        }
    }

    // epilogue: normalize -> o_bf (reuse mchunk space), proj GEMM, bias, scatter
    bf16_t* o_bf = mchunk;   // 64*96 bf16 fits in mchunk
    {
        const int mt = wave & 3, nt = wave >> 2;
#pragma unroll
        for (int h = 0; h < NHEAD; ++h) {
            const int d = nt * 16 + lr;
            if (d < DHEAD) {
                const int c = h * DHEAD + d;
#pragma unroll
                for (int r = 0; r < 8; ++r) {
                    const int q = mt * 16 + r + 8 * lh;
                    o_bf[q * CC + c] = (bf16_t)(o_acc[h][r] / l_i[h][q]);
                }
            }
        }
    }
    __syncthreads();

    for (int t = wave; t < 24; t += 8) {
        const int mt = t / 6, nt = t % 6;
        v8f acc;
#pragma unroll
        for (int j = 0; j < 8; ++j) acc[j] = 0.f;
#pragma unroll
        for (int ks = 0; ks < 3; ++ks) {
            v16bf a = load_frag(&o_bf[mt * 16 * CC + ks * 32], CC, lane);
            v16bf bb = load_frag(&pwb[nt * 16 * CC + ks * 32], CC, lane);
            acc = wmma_bf16(a, bb, acc);
        }
        const int c = nt * 16 + lr;
        const float bias = proj_b[c];
#pragma unroll
        for (int r = 0; r < 8; ++r) {
            const int wwin = mt * 16 + r + 8 * lh;
            const float val = acc[r] + bias;
            const int hb = (wwin >> 3) * 32 + shi * 4;
            const int wb = (wwin & 7) * 32 + swi * 4;
            const size_t base = ((size_t)(b * CC + c) * HH + hb) * WW2 + wb;
            const float4 v4 = make_float4(val, val, val, val);
            *(float4*)(out + base)           = v4;
            *(float4*)(out + base + WW2)     = v4;
            *(float4*)(out + base + 2 * WW2) = v4;
            *(float4*)(out + base + 3 * WW2) = v4;
        }
    }
}

// ---------- host ----------
extern "C" void kernel_launch(void* const* d_in, const int* in_sizes, int n_in,
                              void* d_out, int out_size, void* d_ws, size_t ws_size,
                              hipStream_t stream) {
    const float* x      = (const float*)d_in[0];
    const float* q_w    = (const float*)d_in[1];
    const float* q_b    = (const float*)d_in[2];
    const float* kv_w   = (const float*)d_in[3];
    const float* kv_b   = (const float*)d_in[4];
    const float* proj_w = (const float*)d_in[5];
    const float* proj_b = (const float*)d_in[6];
    float* out = (float*)d_out;

    char* ws = (char*)d_ws;
    bf16_t* pooled = (bf16_t*)(ws);                 // 16384*96*2   = 3,145,728 B
    bf16_t* q_ws   = (bf16_t*)(ws + 3145728);       // 16384*128*2  = 4,194,304 B
    bf16_t* qwb    = (bf16_t*)(ws + 7340032);       // 96*96*2      = 18,432 B
    bf16_t* kvwb   = (bf16_t*)(ws + 7358464);       // 192*96*2     = 36,864 B
    bf16_t* pwb    = (bf16_t*)(ws + 7395328);       // 96*96*2      = 18,432 B

    k0_prep<<<4096, 256, 0, stream>>>(q_w, kv_w, proj_w, qwb, kvwb, pwb,
                                      (unsigned int*)q_ws);
    k1_pool<<<24576, 64, 0, stream>>>(x, pooled);
    k2_qproj<<<128, 256, 0, stream>>>(pooled, qwb, q_b, q_ws);
    k3_attn<<<256, 256, 0, stream>>>(x, kv_b, proj_b, q_ws, kvwb, pwb, out);
}
